// Block_34127810134631
// MI455X (gfx1250) — matrix operand order repaired and checked by reading, not compiled, not run on hardware
//
#include <hip/hip_runtime.h>
#include <hip/hip_bf16.h>

// ---------------------------------------------------------------------------
// MI455X (gfx1250) implementation. wave32. All GEMM-shaped math goes through
// v_wmma_f32_16x16x32_f16; attention is fused flash-style. Guard-free LDS
// staging via padded buffers (rows -> 8256, adapter K 48 -> 64 zero-padded);
// global_prefetch_b8 for the next K tile. Workspace requirement: ~145 MB.
// ---------------------------------------------------------------------------

typedef __attribute__((ext_vector_type(16))) _Float16     v16h;
typedef __attribute__((ext_vector_type(8)))  float        v8f;
typedef __attribute__((ext_vector_type(4)))  unsigned int u32x4;

union V16H { v16h v; u32x4 u[2]; _Float16 h[16]; };

#define C_     384
#define HEADS  6
#define DH     64
#define BATCH  8
#define GP1    1025            // GP + 1 (includes cls token)
#define NROWS  (BATCH * GP1)   // 8200
#define MPAD   8256            // NROWS padded to multiple of 64 (GEMM row tile)
#define NFEAT  (BATCH * 1024)  // 8192 (already multiple of 64)
#define MSEG   1024
#define NSUP   6
#define GSEG   (BATCH * 192)   // 1536
#define KADP   64              // adapter bottleneck 48 padded to 64

__device__ __forceinline__ float wsum32(float v) {
#pragma unroll
  for (int o = 16; o >= 1; o >>= 1) v += __shfl_xor(v, o, 32);
  return v;
}
__device__ __forceinline__ float hmax16(float v) {
#pragma unroll
  for (int o = 8; o >= 1; o >>= 1) v = fmaxf(v, __shfl_xor(v, o, 16));
  return v;
}
__device__ __forceinline__ float hsum16(float v) {
#pragma unroll
  for (int o = 8; o >= 1; o >>= 1) v += __shfl_xor(v, o, 16);
  return v;
}

// ---------------------------------------------------------------------------
// helpers: casts / zero fill
// ---------------------------------------------------------------------------
__global__ void k_cast_f16(const float* __restrict__ src, _Float16* __restrict__ dst, int n) {
  int i = blockIdx.x * 256 + threadIdx.x;
  if (i < n) dst[i] = (_Float16)src[i];
}

// dst[r*dstK + k] = (r<srcRows && k<srcK) ? src[r*srcK+k] : 0   (zero padding)
__global__ void k_cast_pad(const float* __restrict__ src, _Float16* __restrict__ dst,
                           int srcRows, int dstRows, int srcK, int dstK) {
  int i = blockIdx.x * 256 + threadIdx.x;
  if (i >= dstRows * dstK) return;
  int r = i / dstK, k = i - r * dstK;
  dst[i] = (r < srcRows && k < srcK) ? (_Float16)src[(size_t)r * srcK + k] : (_Float16)0.0f;
}

__global__ void k_zero_f16(_Float16* __restrict__ dst, int n) {
  int i = blockIdx.x * 256 + threadIdx.x;
  if (i < n) dst[i] = (_Float16)0.0f;
}

// ---------------------------------------------------------------------------
// LayerNorm (C=384) -> f16, one wave per row
// ---------------------------------------------------------------------------
__global__ __launch_bounds__(128)
void k_ln_f16(const float* __restrict__ in, const float* __restrict__ g,
              const float* __restrict__ bta, _Float16* __restrict__ out, int rows) {
  int wv = threadIdx.x >> 5, lane = threadIdx.x & 31;
  int row = blockIdx.x * 4 + wv;
  if (row >= rows) return;
  const float* p = in + (size_t)row * C_;
  float vals[12];
  float s = 0.f;
#pragma unroll
  for (int j = 0; j < 12; ++j) { vals[j] = p[lane + 32 * j]; s += vals[j]; }
  s = wsum32(s);
  float mean = s * (1.0f / 384.0f);
  float q = 0.f;
#pragma unroll
  for (int j = 0; j < 12; ++j) { float d = vals[j] - mean; q += d * d; }
  q = wsum32(q);
  float rstd = rsqrtf(q * (1.0f / 384.0f) + 1e-5f);
#pragma unroll
  for (int j = 0; j < 12; ++j) {
    int c = lane + 32 * j;
    out[(size_t)row * C_ + c] = (_Float16)((vals[j] - mean) * rstd * g[c] + bta[c]);
  }
}

// ---------------------------------------------------------------------------
// WMMA GEMM: out = epilogue(A[Mr x K] @ W[Nc x K]^T + bias)
// 256 threads = 8 waves, 64x64 block tile, BK=64 (two WMMA K-steps / stage).
// GUARD-FREE staging: caller guarantees A has >= rowTile+64 rows allocated,
// W has >= colTile+64 rows allocated, K % 64 == 0, all 16B aligned.
// Epilogue: v = act(acc+bias); v *= accScale; v *= rowmask[row];
//           v += res1; v += res2; store f32 and/or f16 (stores are guarded).
// ---------------------------------------------------------------------------
__global__ __launch_bounds__(256)
void k_gemm_f16(const _Float16* __restrict__ A, int lda,
                const _Float16* __restrict__ W, int ldw,
                const float* __restrict__ bias,
                float* __restrict__ outF, _Float16* __restrict__ outH,
                const float* __restrict__ res1, const float* __restrict__ res2,
                const float* __restrict__ rowmask,
                int Mr, int Nc, int K, int ldo, int act, float accScale) {
  __shared__ _Float16 As[64][72];
  __shared__ _Float16 Bs[64][72];
  const int tid  = threadIdx.x;
  const int lane = tid & 31;
  const int wv   = tid >> 5;
  const int wr   = wv & 3;
  const int wc   = wv >> 2;
  const int hf   = lane >> 4;
  const int l16  = lane & 15;
  const int rowTile = blockIdx.y * 64;
  const int colTile = blockIdx.x * 64;

  v8f acc0 = {0, 0, 0, 0, 0, 0, 0, 0};
  v8f acc1 = {0, 0, 0, 0, 0, 0, 0, 0};

  for (int k0 = 0; k0 < K; k0 += 64) {
    // stage A tile (64x64) and W tile (64x64): 2 chunks of 8 halves / thread
#pragma unroll
    for (int c = 0; c < 2; ++c) {
      int idx = tid + c * 256;
      int r = idx >> 3, kc = (idx & 7) << 3;
      *(u32x4*)&As[r][kc] = *(const u32x4*)(A + (size_t)(rowTile + r) * lda + k0 + kc);
      *(u32x4*)&Bs[r][kc] = *(const u32x4*)(W + (size_t)(colTile + r) * ldw + k0 + kc);
    }
    if (k0 + 64 < K) {  // prefetch next K tile (global_prefetch_b8)
      int r = tid >> 3, kc = (tid & 7) << 3;
      __builtin_prefetch(A + (size_t)(rowTile + r) * lda + k0 + 64 + kc, 0, 3);
      __builtin_prefetch(W + (size_t)(colTile + r) * ldw + k0 + 64 + kc, 0, 3);
    }
    __syncthreads();
    // Fragments: A lane m<16 holds row m, K {0..15};
    // lanes 16-31 K {16..31}. B lane half holds K{0..15}/{16..31} of
    // column lane%16 (W row n is B column n since out = A @ W^T).
#pragma unroll
    for (int ks = 0; ks < 2; ++ks) {
      V16H a, b0, b1;
      int row = wr * 16 + l16;
      int ab = ks * 32 + hf * 16;
      a.u[0] = *(const u32x4*)&As[row][ab];
      a.u[1] = *(const u32x4*)&As[row][ab + 8];
      int n0 = wc * 32 + l16;
      int bb = ks * 32 + hf * 16;
      b0.u[0] = *(const u32x4*)&Bs[n0][bb];
      b0.u[1] = *(const u32x4*)&Bs[n0][bb + 8];
      b1.u[0] = *(const u32x4*)&Bs[n0 + 16][bb];
      b1.u[1] = *(const u32x4*)&Bs[n0 + 16][bb + 8];
      acc0 = __builtin_amdgcn_wmma_f32_16x16x32_f16(false, a.v, false, b0.v, (short)0, acc0, false, false);
      acc1 = __builtin_amdgcn_wmma_f32_16x16x32_f16(false, a.v, false, b1.v, (short)0, acc1, false, false);
    }
    __syncthreads();
  }

  // Epilogue. C layout: VGPR r -> row r + 8*(lane/16), col = lane%16.
#pragma unroll
  for (int cb = 0; cb < 2; ++cb) {
    v8f acc = cb ? acc1 : acc0;
    int colg = colTile + wc * 32 + cb * 16 + l16;
    if (colg >= Nc) continue;
    float bv = bias ? bias[colg] : 0.0f;
#pragma unroll
    for (int rr = 0; rr < 8; ++rr) {
      int rowg = rowTile + wr * 16 + hf * 8 + rr;
      if (rowg >= Mr) continue;
      float v = acc[rr] + bv;
      if (act == 1) v = v / (1.0f + __expf(-1.702f * v));  // quick_gelu
      v *= accScale;
      if (rowmask) v *= rowmask[rowg];
      size_t o = (size_t)rowg * ldo + colg;
      if (res1) v += res1[o];
      if (res2) v += res2[o];
      if (outF) outF[o] = v;
      if (outH) outH[o] = (_Float16)v;
    }
  }
}

// ---------------------------------------------------------------------------
// Fused flash attention: one wave per (batch, head, 16-query tile).
// qkv layout: [row(8200)][1152] f16, q=0..383, k=384..767, v=768..1151.
// Branchless staging: clamped-address b128 loads + vector select.
// ---------------------------------------------------------------------------
__global__ __launch_bounds__(32)
void k_attn(const _Float16* __restrict__ qkv, _Float16* __restrict__ o16) {
  __shared__ _Float16 Qs[16][72];   // 16 queries x 64 dh
  __shared__ _Float16 Ks[32][72];   // 32 keys    x 64 dh
  __shared__ _Float16 Vt[64][40];   // dh x 32 keys (transposed for B frags)
  __shared__ _Float16 Ps[16][40];   // 16 queries x 32 keys (P tile, f16)

  const int qt = blockIdx.x;
  const int h  = blockIdx.y;
  const int b  = blockIdx.z;
  const int lane = threadIdx.x;
  const int hf = lane >> 4, l16 = lane & 15;
  const u32x4 zv = {0, 0, 0, 0};

  { // Q tile: lane -> row l16, 32 halves at dh offset hf*32 (zero OOB rows)
    int q = qt * 16 + l16;
    bool ok = q < GP1;
    const _Float16* src = qkv + ((size_t)(b * GP1 + (ok ? q : 0))) * 1152 + h * DH + hf * 32;
#pragma unroll
    for (int c = 0; c < 4; ++c) {
      u32x4 t = *(const u32x4*)(src + c * 8);
      *(u32x4*)&Qs[l16][hf * 32 + c * 8] = ok ? t : zv;
    }
  }
  __syncthreads();
  V16H aQ0, aQ1;  // A frags for dh K-steps 0..31 / 32..63 (constant over chunks)
  aQ0.u[0] = *(const u32x4*)&Qs[l16][hf * 16];
  aQ0.u[1] = *(const u32x4*)&Qs[l16][hf * 16 + 8];
  aQ1.u[0] = *(const u32x4*)&Qs[l16][32 + hf * 16];
  aQ1.u[1] = *(const u32x4*)&Qs[l16][32 + hf * 16 + 8];

  float m_i[8], l_i[8];
  v8f vz = {0, 0, 0, 0, 0, 0, 0, 0};
  v8f accO[4];
#pragma unroll
  for (int t = 0; t < 8; ++t) { m_i[t] = -3.0e38f; l_i[t] = 0.0f; }
#pragma unroll
  for (int ob = 0; ob < 4; ++ob) accO[ob] = vz;

  const int nch = (GP1 + 31) / 32;  // 33 key chunks of 32
  for (int ch = 0; ch < nch; ++ch) {
    int kb = ch * 32;
    { // K rows + transposed V for this chunk; lane -> key row 'lane'
      int kr = kb + lane;
      bool ok = kr < GP1;
      const _Float16* sK = qkv + ((size_t)(b * GP1 + (ok ? kr : 0))) * 1152 + C_ + h * DH;
      const _Float16* sV = qkv + ((size_t)(b * GP1 + (ok ? kr : 0))) * 1152 + 2 * C_ + h * DH;
#pragma unroll
      for (int c = 0; c < 8; ++c) {
        u32x4 t = *(const u32x4*)(sK + c * 8);
        *(u32x4*)&Ks[lane][c * 8] = ok ? t : zv;
      }
#pragma unroll
      for (int d = 0; d < DH; ++d) Vt[d][lane] = ok ? sV[d] : (_Float16)0.0f;
    }
    __syncthreads();

    // S = Q K^T (16x32 = two 16x16 frags, 2 dh K-steps each)
    v8f s0 = vz, s1 = vz;
    {
      V16H bk;
      int n = l16;
      bk.u[0] = *(const u32x4*)&Ks[n][hf * 16];
      bk.u[1] = *(const u32x4*)&Ks[n][hf * 16 + 8];
      s0 = __builtin_amdgcn_wmma_f32_16x16x32_f16(false, aQ0.v, false, bk.v, (short)0, s0, false, false);
      bk.u[0] = *(const u32x4*)&Ks[n][32 + hf * 16];
      bk.u[1] = *(const u32x4*)&Ks[n][32 + hf * 16 + 8];
      s0 = __builtin_amdgcn_wmma_f32_16x16x32_f16(false, aQ1.v, false, bk.v, (short)0, s0, false, false);
      n = 16 + l16;
      bk.u[0] = *(const u32x4*)&Ks[n][hf * 16];
      bk.u[1] = *(const u32x4*)&Ks[n][hf * 16 + 8];
      s1 = __builtin_amdgcn_wmma_f32_16x16x32_f16(false, aQ0.v, false, bk.v, (short)0, s1, false, false);
      bk.u[0] = *(const u32x4*)&Ks[n][32 + hf * 16];
      bk.u[1] = *(const u32x4*)&Ks[n][32 + hf * 16 + 8];
      s1 = __builtin_amdgcn_wmma_f32_16x16x32_f16(false, aQ1.v, false, bk.v, (short)0, s1, false, false);
    }
    bool v0 = (kb + l16) < GP1;
    bool v1 = (kb + 16 + l16) < GP1;
    // Online softmax per query row: a row's 16 key-columns live exactly in
    // one 16-lane half at accumulator element rr -> width-16 shuffle reduce.
#pragma unroll
    for (int rr = 0; rr < 8; ++rr) {
      float x0 = v0 ? s0[rr] * 0.125f : -3.0e38f;  // dh^-0.5 = 1/8
      float x1 = v1 ? s1[rr] * 0.125f : -3.0e38f;
      float tm = hmax16(fmaxf(x0, x1));
      float mn = fmaxf(m_i[rr], tm);
      float corr = __expf(m_i[rr] - mn);
      float p0 = __expf(x0 - mn);
      float p1 = __expf(x1 - mn);
      float psum = hsum16(p0 + p1);
      l_i[rr] = l_i[rr] * corr + psum;
      m_i[rr] = mn;
#pragma unroll
      for (int ob = 0; ob < 4; ++ob) accO[ob][rr] = accO[ob][rr] * corr;
      Ps[hf * 8 + rr][l16] = (_Float16)p0;        // C-layout -> LDS
      Ps[hf * 8 + rr][16 + l16] = (_Float16)p1;
    }
    __syncthreads();
    V16H aP;                                       // LDS -> A-layout
    aP.u[0] = *(const u32x4*)&Ps[l16][hf * 16];
    aP.u[1] = *(const u32x4*)&Ps[l16][hf * 16 + 8];
#pragma unroll
    for (int ob = 0; ob < 4; ++ob) {               // O += P @ V (4 dh blocks)
      V16H bv;
      bv.u[0] = *(const u32x4*)&Vt[ob * 16 + l16][hf * 16];
      bv.u[1] = *(const u32x4*)&Vt[ob * 16 + l16][hf * 16 + 8];
      accO[ob] = __builtin_amdgcn_wmma_f32_16x16x32_f16(false, aP.v, false, bv.v, (short)0, accO[ob], false, false);
    }
    __syncthreads();
  }

#pragma unroll
  for (int ob = 0; ob < 4; ++ob) {
#pragma unroll
    for (int rr = 0; rr < 8; ++rr) {
      int q = qt * 16 + hf * 8 + rr;
      if (q < GP1) {
        float v = accO[ob][rr] / fmaxf(l_i[rr], 1e-20f);
        o16[((size_t)(b * GP1 + q)) * C_ + h * DH + ob * 16 + l16] = (_Float16)v;
      }
    }
  }
}

// ---------------------------------------------------------------------------
// Segment max+mean over sorted clusters (seg_id ascending): block = segment,
// thread = channel; binary-search the contiguous row range.
// ---------------------------------------------------------------------------
__global__ __launch_bounds__(384)
void k_pool_sorted(const float* __restrict__ feat, const int* __restrict__ order,
                   const int* __restrict__ segid, float* __restrict__ pooled) {
  const int s = blockIdx.x;
  const int c = threadIdx.x;
  int lo = 0, hi = NFEAT;
  while (lo < hi) { int mid = (lo + hi) >> 1; if (segid[mid] < s) lo = mid + 1; else hi = mid; }
  int start = lo;
  hi = NFEAT;
  while (lo < hi) { int mid = (lo + hi) >> 1; if (segid[mid] <= s) lo = mid + 1; else hi = mid; }
  int end = lo;
  float sum = 0.f, mx = -3.0e38f;
  for (int rr = start; rr < end; ++rr) {
    int row = order[rr];
    float v = feat[(size_t)row * C_ + c];
    sum += v;
    mx = fmaxf(mx, v);
  }
  int cnt = end - start;
  float mean = sum / fmaxf((float)cnt, 1.0f);
  pooled[(size_t)s * C_ + c] = ((cnt > 0) ? mx : 0.0f) + mean;
}

// ---------------------------------------------------------------------------
// Segment max+mean over unsorted flat_grid_index: block = segment, thread =
// channel; scan rows with LDS-staged index cache (deterministic, no atomics).
// ---------------------------------------------------------------------------
__global__ __launch_bounds__(384)
void k_pool_scan(const float* __restrict__ fx, const int* __restrict__ idx,
                 float* __restrict__ pooled) {
  __shared__ int cache[384];
  const int s = blockIdx.x;
  const int c = threadIdx.x;
  float sum = 0.f, mx = -3.0e38f;
  int cnt = 0;
  for (int base = 0; base < NFEAT; base += 384) {
    int t = base + c;
    cache[c] = (t < NFEAT) ? idx[t] : -1;
    __syncthreads();
    int lim = NFEAT - base; if (lim > 384) lim = 384;
    for (int j = 0; j < lim; ++j) {
      if (cache[j] == s) {
        float v = fx[(size_t)(base + j) * C_ + c];
        sum += v;
        mx = fmaxf(mx, v);
        ++cnt;
      }
    }
    __syncthreads();
  }
  float mean = sum / fmaxf((float)cnt, 1.0f);
  pooled[(size_t)s * C_ + c] = ((cnt > 0) ? mx : 0.0f) + mean;
}

// ---------------------------------------------------------------------------
// BatchNorm stats over rows (per channel), then apply + exact GELU
// ---------------------------------------------------------------------------
__global__ __launch_bounds__(256)
void k_bn_stats(const float* __restrict__ in, int R, float* __restrict__ stats) {
  const int c = blockIdx.x;
  float s = 0.f, sq = 0.f;
  for (int rr = threadIdx.x; rr < R; rr += 256) {
    float v = in[(size_t)rr * C_ + c];
    s += v; sq += v * v;
  }
  __shared__ float sh[256], sh2[256];
  sh[threadIdx.x] = s; sh2[threadIdx.x] = sq;
  __syncthreads();
  for (int o = 128; o > 0; o >>= 1) {
    if (threadIdx.x < o) { sh[threadIdx.x] += sh[threadIdx.x + o]; sh2[threadIdx.x] += sh2[threadIdx.x + o]; }
    __syncthreads();
  }
  if (threadIdx.x == 0) {
    float m = sh[0] / (float)R;
    stats[c] = m;
    stats[C_ + c] = sh2[0] / (float)R - m * m;
  }
}

__global__ void k_bn_apply_gelu(const float* __restrict__ in, const float* __restrict__ stats,
                                const float* __restrict__ g, const float* __restrict__ bta,
                                float* __restrict__ out, int n) {
  int i = blockIdx.x * 256 + threadIdx.x;
  if (i >= n) return;
  int c = i % C_;
  float m = stats[c], v = stats[C_ + c];
  float y = (in[i] - m) * rsqrtf(v + 1e-5f) * g[c] + bta[c];
  out[i] = 0.5f * y * (1.0f + erff(y * 0.70710678f));
}

// ---------------------------------------------------------------------------
// feat = x_cur[:, 1:, :] flattened
// ---------------------------------------------------------------------------
__global__ void k_copy_feat(const float* __restrict__ xc, float* __restrict__ feat) {
  int i = blockIdx.x * 256 + threadIdx.x;
  if (i >= NFEAT * C_) return;
  int row = i / C_, c = i - row * C_;
  int b = row >> 10, gm = row & 1023;
  feat[i] = xc[((size_t)(b * GP1 + 1 + gm)) * C_ + c];
}

// ---------------------------------------------------------------------------
// Final: cosine-sim weighted combine + residual; writes full (B,1025,C) out.
// One wave per output row.
// ---------------------------------------------------------------------------
__global__ __launch_bounds__(256)
void k_final(const float* __restrict__ x_cur, const float* __restrict__ x3dseg,
             const float* __restrict__ d2seg, const int* __restrict__ cluster,
             const int* __restrict__ fgi, float* __restrict__ out) {
  int wv = threadIdx.x >> 5, lane = threadIdx.x & 31;
  int rowx = blockIdx.x * 8 + wv;
  if (rowx >= NROWS) return;
  int b = rowx / GP1, g = rowx % GP1;
  const float* src = x_cur + (size_t)rowx * C_;
  float* dst = out + (size_t)rowx * C_;
  if (g == 0) {  // cls token passthrough
    for (int c = lane; c < C_; c += 32) dst[c] = src[c];
    return;
  }
  int n = b * 1024 + (g - 1);
  float x3[12];
  int cl = cluster[n];
  const float* p3 = x3dseg + (size_t)cl * C_;
  float nq = 0.f;
#pragma unroll
  for (int j = 0; j < 12; ++j) { x3[j] = p3[lane + 32 * j]; nq += x3[j] * x3[j]; }
  nq = wsum32(nq);
  float na = fmaxf(sqrtf(nq), 1e-8f);
  float sims[NSUP]; int segs[NSUP];
  float ssum = 0.f;
#pragma unroll
  for (int s = 0; s < NSUP; ++s) {
    int sg = fgi[s * NFEAT + n];
    segs[s] = sg;
    const float* pd = d2seg + ((size_t)s * GSEG + sg) * C_;
    float dot = 0.f, nb = 0.f;
#pragma unroll
    for (int j = 0; j < 12; ++j) { float d = pd[lane + 32 * j]; dot += d * x3[j]; nb += d * d; }
    dot = wsum32(dot); nb = wsum32(nb);
    float cosv = dot / (na * fmaxf(sqrtf(nb), 1e-8f));
    sims[s] = (cosv + 1.0f) * 0.5f;
    ssum += sims[s];
  }
  float inv = 1.0f / ssum;
#pragma unroll
  for (int j = 0; j < 12; ++j) {
    int c = lane + 32 * j;
    float acc = 0.f;
#pragma unroll
    for (int s = 0; s < NSUP; ++s)
      acc += sims[s] * inv * d2seg[((size_t)s * GSEG + segs[s]) * C_ + c];
    dst[c] = src[c] + 0.3f * acc;  // COEF_PRO
  }
}

// ---------------------------------------------------------------------------
// Host launch sequence (graph-capture safe: only kernel launches + async d2d)
// ---------------------------------------------------------------------------
extern "C" void kernel_launch(void* const* d_in, const int* in_sizes, int n_in,
                              void* d_out, int out_size, void* d_ws, size_t ws_size,
                              hipStream_t stream) {
  (void)in_sizes; (void)n_in; (void)out_size; (void)ws_size;

  const float* x      = (const float*)d_in[0];
  const float* mask   = (const float*)d_in[1];
  const float* ln1_g  = (const float*)d_in[2];
  const float* ln1_b  = (const float*)d_in[3];
  const float* ln2_g  = (const float*)d_in[4];
  const float* ln2_b  = (const float*)d_in[5];
  const float* w_in   = (const float*)d_in[6];
  const float* b_in   = (const float*)d_in[7];
  const float* w_out  = (const float*)d_in[8];
  const float* b_out  = (const float*)d_in[9];
  const float* w_fc   = (const float*)d_in[10];
  const float* b_fc   = (const float*)d_in[11];
  const float* w_proj = (const float*)d_in[12];
  const float* b_proj = (const float*)d_in[13];
  const float* wa1    = (const float*)d_in[14];
  const float* ba1    = (const float*)d_in[15];
  const float* wa2    = (const float*)d_in[16];
  const float* ba2    = (const float*)d_in[17];
  const float* g3     = (const float*)d_in[18];
  const float* b3     = (const float*)d_in[19];
  const float* g1d    = (const float*)d_in[20];
  const float* b1d    = (const float*)d_in[21];
  const float* gn3    = (const float*)d_in[22];
  const float* bn3    = (const float*)d_in[23];
  const float* w_attn1 = (const float*)d_in[24];
  const float* b_attn1 = (const float*)d_in[25];
  const int* order    = (const int*)d_in[26];
  const int* seg_id   = (const int*)d_in[27];
  const int* cluster  = (const int*)d_in[28];
  const int* fgi      = (const int*)d_in[29];

  size_t off = 0;
  auto alloc = [&](size_t bytes) -> void* {
    off = (off + 255) & ~(size_t)255;
    void* p = (char*)d_ws + off;
    off += bytes;
    return p;
  };

  // f16 weights (N padded to 64 rows / K padded to 64 where needed)
  _Float16* w_in16   = (_Float16*)alloc((size_t)1152 * 384 * 2);
  _Float16* w_out16  = (_Float16*)alloc((size_t)384 * 384 * 2);
  _Float16* w_fc16   = (_Float16*)alloc((size_t)1536 * 384 * 2);
  _Float16* w_proj16 = (_Float16*)alloc((size_t)384 * 1536 * 2);
  _Float16* wa116    = (_Float16*)alloc((size_t)64 * 384 * 2);     // 48->64 rows, zero pad
  _Float16* wa216    = (_Float16*)alloc((size_t)384 * KADP * 2);   // K 48->64, zero pad
  _Float16* wattn16  = (_Float16*)alloc((size_t)NSUP * 384 * 384 * 2);
  // activations (GEMM A inputs row-padded to MPAD; pad rows never stored)
  float*    x_cur  = (float*)   alloc((size_t)NROWS * C_ * 4);
  _Float16* ln16   = (_Float16*)alloc((size_t)MPAD * C_ * 2);
  _Float16* qkv16  = (_Float16*)alloc((size_t)NROWS * 1152 * 2);
  _Float16* o16    = (_Float16*)alloc((size_t)MPAD * C_ * 2);
  _Float16* h16    = (_Float16*)alloc((size_t)MPAD * 1536 * 2);
  float*    xffn   = (float*)   alloc((size_t)NROWS * C_ * 4);
  _Float16* xffn16 = (_Float16*)alloc((size_t)MPAD * C_ * 2);
  _Float16* a16    = (_Float16*)alloc((size_t)MPAD * KADP * 2);
  float*    feat   = (float*)   alloc((size_t)NFEAT * C_ * 4);
  _Float16* fln16  = (_Float16*)alloc((size_t)NFEAT * C_ * 2);
  float*    fx     = (float*)   alloc((size_t)NFEAT * C_ * 4);
  float*    pooled = (float*)   alloc((size_t)GSEG * C_ * 4);
  float*    stats  = (float*)   alloc((size_t)2 * C_ * 4);
  float*    x3dseg = (float*)   alloc((size_t)MSEG * C_ * 4);
  float*    d2seg  = (float*)   alloc((size_t)NSUP * GSEG * C_ * 4);

  auto castN = [&](const float* s, _Float16* d, int n) {
    k_cast_f16<<<(n + 255) / 256, 256, 0, stream>>>(s, d, n);
  };
  castN(w_in, w_in16, 1152 * 384);
  castN(w_out, w_out16, 384 * 384);
  castN(w_fc, w_fc16, 1536 * 384);
  castN(w_proj, w_proj16, 384 * 1536);
  castN(w_attn1, wattn16, NSUP * 384 * 384);
  // zero-padded adapter weights
  k_cast_pad<<<(64 * 384 + 255) / 256, 256, 0, stream>>>(wa1, wa116, 48, 64, 384, 384);
  k_cast_pad<<<(384 * KADP + 255) / 256, 256, 0, stream>>>(wa2, wa216, 384, 384, 48, KADP);

  hipMemcpyAsync(x_cur, x, (size_t)NROWS * C_ * 4, hipMemcpyDeviceToDevice, stream);

  auto gemmGrid = [](int Mr, int Nc) { return dim3((Nc + 63) / 64, (Mr + 63) / 64); };

  // ---- attention block ----
  k_ln_f16<<<(NROWS + 3) / 4, 128, 0, stream>>>(x_cur, ln1_g, ln1_b, ln16, NROWS);
  k_gemm_f16<<<gemmGrid(NROWS, 1152), 256, 0, stream>>>(
      ln16, 384, w_in16, 384, b_in, nullptr, qkv16, nullptr, nullptr, nullptr,
      NROWS, 1152, 384, 1152, 0, 1.0f);
  k_attn<<<dim3((GP1 + 15) / 16, HEADS, BATCH), 32, 0, stream>>>(qkv16, o16);
  k_gemm_f16<<<gemmGrid(NROWS, 384), 256, 0, stream>>>(
      o16, 384, w_out16, 384, b_out, x_cur, nullptr, x_cur, nullptr, nullptr,
      NROWS, 384, 384, 384, 0, 1.0f);

  // ---- FFN + adapter ----
  k_ln_f16<<<(NROWS + 3) / 4, 128, 0, stream>>>(x_cur, ln2_g, ln2_b, ln16, NROWS);
  k_gemm_f16<<<gemmGrid(NROWS, 1536), 256, 0, stream>>>(
      ln16, 384, w_fc16, 384, b_fc, nullptr, h16, nullptr, nullptr, nullptr,
      NROWS, 1536, 384, 1536, 1, 1.0f);
  k_gemm_f16<<<gemmGrid(NROWS, 384), 256, 0, stream>>>(
      h16, 1536, w_proj16, 1536, b_proj, xffn, xffn16, nullptr, nullptr, nullptr,
      NROWS, 384, 1536, 384, 0, 1.0f);
  k_zero_f16<<<(MPAD * KADP + 255) / 256, 256, 0, stream>>>(a16, MPAD * KADP);
  k_gemm_f16<<<gemmGrid(NROWS, 48), 256, 0, stream>>>(
      xffn16, 384, wa116, 384, ba1, nullptr, a16, nullptr, nullptr, nullptr,
      NROWS, 48, 384, KADP, 1, 1.0f);
  k_gemm_f16<<<gemmGrid(NROWS, 384), 256, 0, stream>>>(
      a16, KADP, wa216, KADP, ba2, x_cur, nullptr, x_cur, xffn, nullptr,
      NROWS, 384, KADP, 384, 0, 0.5f);  // x += x_ffn + 0.5*adapter

  // ---- cluster pooling branch ----
  k_copy_feat<<<(NFEAT * C_ + 255) / 256, 256, 0, stream>>>(x_cur, feat);
  k_pool_sorted<<<MSEG, 384, 0, stream>>>(feat, order, seg_id, pooled);
  k_bn_stats<<<C_, 256, 0, stream>>>(pooled, MSEG, stats);
  k_bn_apply_gelu<<<(MSEG * C_ + 255) / 256, 256, 0, stream>>>(
      pooled, stats, g3, b3, x3dseg, MSEG * C_);

  // ---- NS grid branches ----
  for (int i = 0; i < NSUP; ++i) {
    k_ln_f16<<<(NFEAT + 3) / 4, 128, 0, stream>>>(
        feat, gn3 + i * C_, bn3 + i * C_, fln16, NFEAT);
    k_gemm_f16<<<gemmGrid(NFEAT, 384), 256, 0, stream>>>(
        fln16, 384, wattn16 + (size_t)i * 384 * 384, 384, b_attn1 + i * C_,
        fx, nullptr, feat, nullptr, mask + (size_t)i * NFEAT,
        NFEAT, 384, 384, 384, 0, 1.0f);
    k_pool_scan<<<GSEG, 384, 0, stream>>>(fx, fgi + i * NFEAT, pooled);
    k_bn_stats<<<C_, 256, 0, stream>>>(pooled, GSEG, stats);
    k_bn_apply_gelu<<<(GSEG * C_ + 255) / 256, 256, 0, stream>>>(
        pooled, stats, g1d + i * C_, b1d + i * C_,
        d2seg + (size_t)i * GSEG * C_, GSEG * C_);
  }

  // ---- final combine ----
  k_final<<<(NROWS + 7) / 8, 256, 0, stream>>>(
      x_cur, x3dseg, d2seg, cluster, fgi, (float*)d_out);
}
